// GCCF_52716428591247
// MI455X (gfx1250) — compile-verified, matrix operand order checked by reading, not run on hardware
//
#include <hip/hip_runtime.h>
#include <hip/hip_bf16.h>

// ---------------------------------------------------------------------------
// GCCF forward on MI455X (gfx1250).
// Heavy work = 6 dense GEMMs  adj[R,K] @ feat[K,64]  (K up to 20000).
// HBM-bound: 4.8 GB adjacency stream @ 23.3 TB/s ~= 206 us floor.
// bf16 WMMA (V_WMMA_F32_16X16X32_BF16) keeps matrix throughput far above the
// ~745 TFLOP/s needed to stay bandwidth-bound (fp32 16x16x4 WMMA would not).
// 16 rows/wave (4 C tiles = 32 VGPRs): low register pressure -> the compiler
// clusters the 4 A-tile b128 loads per chunk (s_clause) with staggered
// s_wait_loadcnt, no spills, and high wave occupancy to hide HBM latency.
// Adjacency loads are non-temporal (TH_NT) so the streamed 800 MB matrices
// don't evict the L2-resident bf16 feature buffers re-read by every wave.
// ---------------------------------------------------------------------------

#define NU 20000
#define NM 10000
#define EDIM 64
#define NL 3
#define NB 8192
#define SLOPE 0.01f

typedef __attribute__((ext_vector_type(16))) __bf16 v16bf;
typedef __attribute__((ext_vector_type(8)))  __bf16 v8bf;
typedef __attribute__((ext_vector_type(8)))  float  v8f;
typedef __attribute__((ext_vector_type(4)))  float  f32x4;

// Build a bf16 A/B fragment from fp32:
// frag[0..7]  = rowp[koff + 0..7]    frag[8..15] = rowp[koff + 16..23]
// koff = (lane<16 ? 0 : 8) -> half-wave K interleave of the CDNA5 16-bit
// 16x32 A layout (and symmetric 32x16 B layout).
__device__ __forceinline__ v16bf frag_from_f32(const float* __restrict__ rowp, int koff) {
    f32x4 x0 = *(const f32x4*)(rowp + koff);
    f32x4 x1 = *(const f32x4*)(rowp + koff + 4);
    f32x4 x2 = *(const f32x4*)(rowp + koff + 16);
    f32x4 x3 = *(const f32x4*)(rowp + koff + 20);
    v16bf a;
#pragma unroll
    for (int i = 0; i < 4; ++i) {
        a[i]      = (__bf16)x0[i];
        a[4 + i]  = (__bf16)x1[i];
        a[8 + i]  = (__bf16)x2[i];
        a[12 + i] = (__bf16)x3[i];
    }
    return a;
}

// Non-temporal variant for the streaming adjacency.
__device__ __forceinline__ v16bf frag_from_f32_nt(const float* __restrict__ rowp, int koff) {
    f32x4 x0 = __builtin_nontemporal_load((const f32x4*)(rowp + koff));
    f32x4 x1 = __builtin_nontemporal_load((const f32x4*)(rowp + koff + 4));
    f32x4 x2 = __builtin_nontemporal_load((const f32x4*)(rowp + koff + 16));
    f32x4 x3 = __builtin_nontemporal_load((const f32x4*)(rowp + koff + 20));
    v16bf a;
#pragma unroll
    for (int i = 0; i < 4; ++i) {
        a[i]      = (__bf16)x0[i];
        a[4 + i]  = (__bf16)x1[i];
        a[8 + i]  = (__bf16)x2[i];
        a[12 + i] = (__bf16)x3[i];
    }
    return a;
}

// Tail (last 16 of K): only low K-group valid; zero the high group.
__device__ __forceinline__ v16bf frag_from_f32_nt_tail(const float* __restrict__ rowp, int koff) {
    f32x4 x0 = __builtin_nontemporal_load((const f32x4*)(rowp + koff));
    f32x4 x1 = __builtin_nontemporal_load((const f32x4*)(rowp + koff + 4));
    v16bf a;
#pragma unroll
    for (int i = 0; i < 4; ++i) {
        a[i]      = (__bf16)x0[i];
        a[4 + i]  = (__bf16)x1[i];
        a[8 + i]  = (__bf16)0.0f;
        a[12 + i] = (__bf16)0.0f;
    }
    return a;
}

// B fragment from pre-transposed bf16 buffer BT[64][Kpad]: two 16B loads.
__device__ __forceinline__ v16bf frag_from_bf16(const __bf16* __restrict__ rowp, int koff) {
    v8bf lo = *(const v8bf*)(rowp + koff);
    v8bf hi = *(const v8bf*)(rowp + koff + 16);
    v16bf b;
#pragma unroll
    for (int i = 0; i < 8; ++i) { b[i] = lo[i]; b[8 + i] = hi[i]; }
    return b;
}

// ---------------------------------------------------------------------------
// feat[K,64] fp32 -> featT[64,Kpad] bf16 (zero-padded K tail).
// ---------------------------------------------------------------------------
__global__ void cvt_transpose_kernel(const float* __restrict__ X, __bf16* __restrict__ XT,
                                     int K, int Kpad) {
    int idx = blockIdx.x * blockDim.x + threadIdx.x;
    int total = EDIM * Kpad;
    if (idx >= total) return;
    int n = idx / Kpad;
    int k = idx - n * Kpad;
    XT[idx] = (k < K) ? (__bf16)X[(size_t)k * EDIM + n] : (__bf16)0.0f;
}

// ---------------------------------------------------------------------------
// S[rows,64] = adj[rows,K] @ BT^T + addv[rows,64]
// One wave per 16-row strip; 4 column tiles (N=64); K step 32.
// ---------------------------------------------------------------------------
__global__ void adj_gemm_kernel(const float* __restrict__ adj, const __bf16* __restrict__ BT,
                                const float* __restrict__ addv, float* __restrict__ S,
                                int rows, int K, int Kpad, int ktail) {
    const int lane = threadIdx.x & 31;
    const int wave = threadIdx.x >> 5;
    const int nstrips = rows >> 4;
    int strip = blockIdx.x * (blockDim.x >> 5) + wave;
    if (strip >= nstrips) strip = nstrips - 1;  // duplicate work, identical stores

    const int row0 = strip << 4;
    const int half = lane >> 4;
    const int ln   = lane & 15;
    const int koff = half << 3;

    const float* arow = adj + (size_t)(row0 + ln) * (size_t)K;
    const __bf16* brow0 = BT + (size_t)(0 * 16 + ln) * (size_t)Kpad;
    const __bf16* brow1 = BT + (size_t)(1 * 16 + ln) * (size_t)Kpad;
    const __bf16* brow2 = BT + (size_t)(2 * 16 + ln) * (size_t)Kpad;
    const __bf16* brow3 = BT + (size_t)(3 * 16 + ln) * (size_t)Kpad;

    v8f acc[4];
#pragma unroll
    for (int t = 0; t < 4; ++t)
#pragma unroll
        for (int i = 0; i < 8; ++i) acc[t][i] = 0.0f;

    const int nfull = K >> 5;
    for (int kc = 0; kc < nfull; ++kc) {
        const int k0 = kc << 5;
        if (k0 + 288 <= K) __builtin_prefetch(arow + k0 + 256, 0, 0);  // global_prefetch_b8
        v16bf a = frag_from_f32_nt(arow + k0, koff);
        v16bf b0 = frag_from_bf16(brow0 + k0, koff);
        v16bf b1 = frag_from_bf16(brow1 + k0, koff);
        v16bf b2 = frag_from_bf16(brow2 + k0, koff);
        v16bf b3 = frag_from_bf16(brow3 + k0, koff);
        acc[0] = __builtin_amdgcn_wmma_f32_16x16x32_bf16(false, a, false, b0, (short)0, acc[0], false, false);
        acc[1] = __builtin_amdgcn_wmma_f32_16x16x32_bf16(false, a, false, b1, (short)0, acc[1], false, false);
        acc[2] = __builtin_amdgcn_wmma_f32_16x16x32_bf16(false, a, false, b2, (short)0, acc[2], false, false);
        acc[3] = __builtin_amdgcn_wmma_f32_16x16x32_bf16(false, a, false, b3, (short)0, acc[3], false, false);
    }
    if (ktail) {  // last 16 of K: high K-group of A zeroed; BT pad rows are zero
        const int k0 = nfull << 5;
        v16bf a = frag_from_f32_nt_tail(arow + k0, koff);
        v16bf b0 = frag_from_bf16(brow0 + k0, koff);
        v16bf b1 = frag_from_bf16(brow1 + k0, koff);
        v16bf b2 = frag_from_bf16(brow2 + k0, koff);
        v16bf b3 = frag_from_bf16(brow3 + k0, koff);
        acc[0] = __builtin_amdgcn_wmma_f32_16x16x32_bf16(false, a, false, b0, (short)0, acc[0], false, false);
        acc[1] = __builtin_amdgcn_wmma_f32_16x16x32_bf16(false, a, false, b1, (short)0, acc[1], false, false);
        acc[2] = __builtin_amdgcn_wmma_f32_16x16x32_bf16(false, a, false, b2, (short)0, acc[2], false, false);
        acc[3] = __builtin_amdgcn_wmma_f32_16x16x32_bf16(false, a, false, b3, (short)0, acc[3], false, false);
    }

    // Epilogue: D layout -> lane (n) = col within tile, VGPR r = row r (+8 upper half)
#pragma unroll
    for (int t = 0; t < 4; ++t) {
        const int col = t * 16 + ln;
#pragma unroll
        for (int r = 0; r < 8; ++r) {
            const int row = row0 + r + half * 8;
            const size_t idx = (size_t)row * EDIM + col;
            S[idx] = acc[t][r] + addv[idx];
        }
    }
}

// ---------------------------------------------------------------------------
// out[rows,64] = leaky_relu(Sin[rows,64] @ W^T + 2*bias), W is [64,64].
// B fragment = rows of W (contiguous fp32), K loop = 2 chunks of 32.
// ---------------------------------------------------------------------------
__global__ void feat_transform_kernel(const float* __restrict__ Sin, const float* __restrict__ W,
                                      const float* __restrict__ bias, float* __restrict__ outp,
                                      int rows) {
    const int lane = threadIdx.x & 31;
    const int wave = threadIdx.x >> 5;
    const int nstrips = rows >> 4;
    int strip = blockIdx.x * (blockDim.x >> 5) + wave;
    if (strip >= nstrips) strip = nstrips - 1;

    const int row0 = strip << 4;
    const int half = lane >> 4;
    const int ln   = lane & 15;
    const int koff = half << 3;

    const float* arow = Sin + (size_t)(row0 + ln) * EDIM;

    v8f acc[4];
#pragma unroll
    for (int t = 0; t < 4; ++t)
#pragma unroll
        for (int i = 0; i < 8; ++i) acc[t][i] = 0.0f;

#pragma unroll
    for (int kc = 0; kc < 2; ++kc) {
        const int k0 = kc << 5;
        v16bf a = frag_from_f32(arow + k0, koff);
#pragma unroll
        for (int t = 0; t < 4; ++t) {
            const float* wrow = W + (size_t)(t * 16 + ln) * EDIM;  // B[k][n] = W[n][k]
            v16bf b = frag_from_f32(wrow + k0, koff);
            acc[t] = __builtin_amdgcn_wmma_f32_16x16x32_bf16(false, a, false, b, (short)0, acc[t], false, false);
        }
    }

#pragma unroll
    for (int t = 0; t < 4; ++t) {
        const int col = t * 16 + ln;
        const float bb = 2.0f * bias[col];
#pragma unroll
        for (int r = 0; r < 8; ++r) {
            const int row = row0 + r + half * 8;
            float x = acc[t][r] + bb;
            outp[(size_t)row * EDIM + col] = (x > 0.0f) ? x : SLOPE * x;
        }
    }
}

// ---------------------------------------------------------------------------
// scores[b] = sum_l sum_e u_l[uid][e]*m_l[mid][e]*outW[l*64+e] + outb
// One wave per interaction; lane covers e and e+32; shfl_xor tree reduce.
// ---------------------------------------------------------------------------
__global__ void score_kernel(const int* __restrict__ uid, const int* __restrict__ mid,
                             const float* __restrict__ u0, const float* __restrict__ u1,
                             const float* __restrict__ u2, const float* __restrict__ u3,
                             const float* __restrict__ m0, const float* __restrict__ m1,
                             const float* __restrict__ m2, const float* __restrict__ m3,
                             const float* __restrict__ outW, const float* __restrict__ outb,
                             float* __restrict__ scores, int nB) {
    const int w = blockIdx.x * (blockDim.x >> 5) + (threadIdx.x >> 5);
    if (w >= nB) return;
    const int lane = threadIdx.x & 31;
    const float* us[4] = {u0, u1, u2, u3};
    const float* ms[4] = {m0, m1, m2, m3};
    const size_t ub = (size_t)uid[w] * EDIM;
    const size_t mb = (size_t)mid[w] * EDIM;
    float acc = 0.0f;
#pragma unroll
    for (int l = 0; l < 4; ++l) {
        acc += us[l][ub + lane]      * ms[l][mb + lane]      * outW[l * EDIM + lane];
        acc += us[l][ub + lane + 32] * ms[l][mb + lane + 32] * outW[l * EDIM + lane + 32];
    }
#pragma unroll
    for (int off = 16; off; off >>= 1) acc += __shfl_xor(acc, off, 32);
    if (lane == 0) scores[w] = acc + outb[0];
}

// ---------------------------------------------------------------------------
extern "C" void kernel_launch(void* const* d_in, const int* in_sizes, int n_in,
                              void* d_out, int out_size, void* d_ws, size_t ws_size,
                              hipStream_t stream) {
    (void)in_sizes; (void)n_in; (void)out_size; (void)ws_size;

    const float* user_adj  = (const float*)d_in[0];
    const float* movie_adj = (const float*)d_in[1];
    const int*   user_id   = (const int*)d_in[2];
    const int*   movie_id  = (const int*)d_in[3];
    const float* user_emb  = (const float*)d_in[4];
    const float* movie_emb = (const float*)d_in[5];
    const float* user_Ws   = (const float*)d_in[6];
    const float* user_bs   = (const float*)d_in[7];
    const float* movie_Ws  = (const float*)d_in[8];
    const float* movie_bs  = (const float*)d_in[9];
    const float* out_W     = (const float*)d_in[10];
    const float* out_b     = (const float*)d_in[11];

    float* scores = (float*)d_out;                    // [B]
    float* u3_out = scores + NB;                      // [NU*E]
    float* m3_out = u3_out + (size_t)NU * EDIM;       // [NM*E]

    const int KPAD_NM = 10016;  // NM padded to 32
    const int KPAD_NU = 20000;  // already multiple of 32

    char* ws = (char*)d_ws;
    float* S_u = (float*)ws;              ws += (size_t)NU * EDIM * 4;
    float* S_m = (float*)ws;              ws += (size_t)NM * EDIM * 4;
    float* U_a = (float*)ws;              ws += (size_t)NU * EDIM * 4;
    float* U_b = (float*)ws;              ws += (size_t)NU * EDIM * 4;
    float* M_a = (float*)ws;              ws += (size_t)NM * EDIM * 4;
    float* M_b = (float*)ws;              ws += (size_t)NM * EDIM * 4;
    __bf16* MbT = (__bf16*)ws;            ws += (size_t)EDIM * KPAD_NM * 2;
    __bf16* UbT = (__bf16*)ws;            // EDIM * KPAD_NU * 2

    const float* uprev[3] = {user_emb, U_a, U_b};
    float*       unext[3] = {U_a, U_b, u3_out};
    const float* mprev[3] = {movie_emb, M_a, M_b};
    float*       mnext[3] = {M_a, M_b, m3_out};

    const dim3 blk(256);  // 8 waves per block for HBM latency hiding
    const int u_strips = NU / 16, m_strips = NM / 16;
    const dim3 ugrid((u_strips + 7) / 8), mgrid((m_strips + 7) / 8);

    for (int l = 0; l < NL; ++l) {
        // features -> bf16 transposed (B-operand layout)
        cvt_transpose_kernel<<<(EDIM * KPAD_NM + 255) / 256, 256, 0, stream>>>(
            mprev[l], MbT, NM, KPAD_NM);
        cvt_transpose_kernel<<<(EDIM * KPAD_NU + 255) / 256, 256, 0, stream>>>(
            uprev[l], UbT, NU, KPAD_NU);
        // S_u = user_adj @ m_prev + u_prev   (K=10000, tail 16)
        adj_gemm_kernel<<<ugrid, blk, 0, stream>>>(
            user_adj, MbT, uprev[l], S_u, NU, NM, KPAD_NM, 16);
        // S_m = movie_adj @ u_prev + m_prev  (K=20000, no tail)
        adj_gemm_kernel<<<mgrid, blk, 0, stream>>>(
            movie_adj, UbT, mprev[l], S_m, NM, NU, KPAD_NU, 0);
        // u_next = lrelu(S_u @ W_u[l]^T + 2 b_u[l]), same for movies
        feat_transform_kernel<<<ugrid, blk, 0, stream>>>(
            S_u, user_Ws + (size_t)l * EDIM * EDIM, user_bs + (size_t)l * EDIM, unext[l], NU);
        feat_transform_kernel<<<mgrid, blk, 0, stream>>>(
            S_m, movie_Ws + (size_t)l * EDIM * EDIM, movie_bs + (size_t)l * EDIM, mnext[l], NM);
    }

    // gather + elementwise product + concat dot with out_W
    score_kernel<<<NB / 8, 256, 0, stream>>>(
        user_id, movie_id,
        user_emb, U_a, U_b, u3_out,
        movie_emb, M_a, M_b, m3_out,
        out_W, out_b, scores, NB);
}
